// InteractionBlock_38062000177170
// MI455X (gfx1250) — matrix operand order repaired and checked by reading, not compiled, not run on hardware
//
#include <hip/hip_runtime.h>
#include <hip/hip_bf16.h>

typedef __attribute__((ext_vector_type(16))) _Float16 v16h;
typedef __attribute__((ext_vector_type(8)))  float    v8f;

#define B_    32
#define M_    256
#define H2_   32
#define OUT_  8
#define PLANE (M_ * M_)                   // 65536
#define TILES_PER_B (M_ * (M_ / 16))      // 4096 = 2^12
#define NTILES (B_ * TILES_PER_B)         // 131072

__device__ __forceinline__ float safelog(float v) {
    return v > 0.0f ? __logf(v) : 0.0f;
}

// One wave handles a tile of 16 consecutive n2 for fixed (b, n1).
// WMMA:  D(16x16) = A(16x32) * B(32x16) + C
//   A = W2 padded to 16 rows (channels), f16, K = hidden dim (32)
//   B = h^T : column n = pair n2b+n, f16
//   C = bias b2 broadcast (channels 8..15 zero)
// D VGPR r, lanes 0..15  -> channel r,   pairs 0..15   (stored)
//          lanes 16..31 -> channel r+8 (zero padding, not stored)
__global__ __launch_bounds__(256) void interaction_wmma_kernel(
    const float* __restrict__ x,     // [B, M, 3]
    const int*   __restrict__ mask,  // [B, M, 1]
    const float* __restrict__ W1,    // [H2, 3]
    const float* __restrict__ b1,    // [H2]
    const float* __restrict__ W2,    // [OUT, H2]
    const float* __restrict__ b2,    // [OUT]
    float*       __restrict__ out)   // [B*OUT, M, M]
{
    const int ln    = threadIdx.x & 31;
    const int half  = ln >> 4;        // 0: K-low half, 1: K-high half
    const int pl    = ln & 15;        // pair-in-tile (N column) / channel row for A
    const int khalf = half << 4;      // 0 or 16: which hidden units this lane owns

    // ---- A operand: W2 in WMMA f16 A-layout (row M = channel = pl) ----
    // element e of lane l maps to K = e + ((l>>4)<<4)
    v16h aW2;
#pragma unroll
    for (int e = 0; e < 16; ++e) {
        const int K = e + (half << 4);
        float w = 0.0f;
        if (pl < OUT_) w = W2[pl * H2_ + K];
        aW2[e] = (_Float16)w;
    }

    // ---- C init: output bias in C/D layout ----
    v8f cbias;
#pragma unroll
    for (int r = 0; r < 8; ++r)
        cbias[r] = (half == 0) ? b2[r] : 0.0f;   // channels 8..15 are padding

    // ---- Per-lane slice of layer-1 weights: hidden units j = khalf + e ----
    float w1m[16], w1d[16], w1k[16], b1r[16];
#pragma unroll
    for (int e = 0; e < 16; ++e) {
        const int j = khalf + e;
        w1m[e] = W1[j * 3 + 0];   // mass_log coeff
        w1d[e] = W1[j * 3 + 1];   // dr_log coeff
        w1k[e] = W1[j * 3 + 2];   // kt_log coeff
        b1r[e] = b1[j];
    }

    const int wid    = (blockIdx.x * blockDim.x + threadIdx.x) >> 5;
    const int nwaves = (gridDim.x * blockDim.x) >> 5;

    for (int t = wid; t < NTILES; t += nwaves) {   // uniform per wave: no divergence
        const int b   = t >> 12;
        const int r2  = t & (TILES_PER_B - 1);
        const int n1  = r2 >> 4;
        const int n2b = (r2 & 15) << 4;
        const int n2  = n2b + pl;                  // lanes l and l+16 share a pair

        // ---- pairwise features (reference math) ----
        const float* xi = x + (size_t)(b * M_ + n1) * 3;
        const float* xj = x + (size_t)(b * M_ + n2) * 3;
        const float eta_i = xi[0], phi_i = xi[1], ept_i = __expf(xi[2]);
        const float eta_j = xj[0], phi_j = xj[1], ept_j = __expf(xj[2]);
        const float me = (float)(mask[b * M_ + n1] * mask[b * M_ + n2]);

        const float de  = eta_i - eta_j;
        const float dp  = phi_i - phi_j;
        const float dr2 = de * de + dp * dp;

        const float dr_log  = 0.5f * safelog(dr2) * me;
        const float dr_lin  = __builtin_sqrtf(dr2) * me;
        const float kt      = fminf(ept_i, ept_j) * dr_lin;
        const float kt_log  = safelog(kt) * me;
        const float ch      = 0.5f * (__expf(de) + __expf(-de));   // cosh(de)
        const float m2      = 2.0f * ept_i * ept_j * (ch - __cosf(dp));
        const float mass_log = safelog(m2) * me;

        // ---- layer 1 (K=3 GEMM on VALU) for this lane's 16 hidden units ----
        v16h bh;
#pragma unroll
        for (int e = 0; e < 16; ++e) {
            float hj = fmaf(mass_log, w1m[e],
                       fmaf(dr_log,   w1d[e],
                       fmaf(kt_log,   w1k[e], b1r[e])));
            hj = hj > 0.0f ? hj : 0.01f * hj;      // leaky_relu
            bh[e] = (_Float16)hj;                  // B layout: lane half owns its K half
        }

        // ---- layer 2 (K=32) on the matrix pipe ----
        v8f d = __builtin_amdgcn_wmma_f32_16x16x32_f16(
            /*neg_a=*/false, aW2, /*neg_b=*/false, bh,
            /*c_mod=*/(short)0, cbias, /*reuse_a=*/false, /*reuse_b=*/false);

        // ---- masked store: channel r plane, 16 contiguous n2 per store ----
        if (half == 0) {
            float* ob = out + (size_t)b * OUT_ * PLANE + (size_t)n1 * M_ + n2;
#pragma unroll
            for (int r = 0; r < OUT_; ++r)
                ob[(size_t)r * PLANE] = d[r] * me;
        }
    }
}

extern "C" void kernel_launch(void* const* d_in, const int* in_sizes, int n_in,
                              void* d_out, int out_size, void* d_ws, size_t ws_size,
                              hipStream_t stream) {
    const float* x    = (const float*)d_in[0];
    const int*   mask = (const int*)  d_in[1];
    const float* W1   = (const float*)d_in[2];
    const float* b1   = (const float*)d_in[3];
    const float* W2   = (const float*)d_in[4];
    const float* b2   = (const float*)d_in[5];
    float*       out  = (float*)d_out;

    // 2048 blocks x 8 waves = 16384 waves; 131072 tiles -> 8 tiles per wave,
    // amortizing the register-resident weight load across the grid-stride loop.
    interaction_wmma_kernel<<<2048, 256, 0, stream>>>(x, mask, W1, b1, W2, b2, out);
}
